// social_stgcnn_1872605741377
// MI455X (gfx1250) — matrix operand order repaired and checked by reading, not compiled, not run on hardware
//
#include <hip/hip_runtime.h>

// ---------------------------------------------------------------------------
// Social-STGCNN forward for MI455X (gfx1250, wave32, WMMA).
// Heavy math (adjacency einsum GEMMs + txpcnn 3x3 convs) runs on
// v_wmma_f32_16x16x32_bf16; everything else is fused elementwise glue.
// Software-pipelined WMMA loops + __launch_bounds__(256) so both fragment
// buffers stay in VGPRs (no scratch spills) and loads overlap the WMMAs.
// ---------------------------------------------------------------------------

typedef __attribute__((ext_vector_type(16))) __bf16 v16bf;
typedef __attribute__((ext_vector_type(8)))  __bf16 v8bf;
typedef __attribute__((ext_vector_type(8)))  float  v8f;

#define N_B    256
#define T_D    8
#define V_D    1024
#define C_OUT  5
#define ROWS   (N_B * C_OUT)          // 1280 GEMM rows (n*5+c)
#define PT     12

// padded txpcnn activation buffer: [N][12][7][1026] bf16, zero halo
#define XC 12
#define XH 7
#define XW 1026
#define CH_STR (XH * XW)              // 7182
#define N_STR  (XC * XH * XW)         // 86184

// workspace layout (bytes)
#define ABT_BYTES  (8ull * 1024 * 1024 * 2)              // 16,777,216  a^T bf16
#define XG_BYTES   (8ull * ROWS * 1024 * 2)              // 20,971,520  gcn out bf16
#define Y_BYTES    ((size_t)N_B * C_OUT * T_D * V_D * 4) // 41,943,040  einsum out f32
#define XBUF_BYTES ((size_t)N_B * N_STR * 2)             // 44,126,208  padded bf16
#define OFF_ABT    0ull
#define OFF_XG     (OFF_ABT + ABT_BYTES)
#define OFF_Y      (OFF_XG + XG_BYTES)
#define OFF_XPONG  0ull                                  // aliases abT+xg+y (dead by then)
#define OFF_XPING  (OFF_Y + Y_BYTES)                     // 79,691,776
#define OFF_WP     (OFF_XPING + XBUF_BYTES)              // 123,817,984 (+20KB)

static __device__ __forceinline__ v16bf ld16bf(const __bf16* p0, const __bf16* p1) {
  union { v16bf v; v8bf h[2]; } u;
  u.h[0] = *(const v8bf*)p0;
  u.h[1] = *(const v8bf*)p1;
  return u.v;
}

// --------------------------- zero helper -----------------------------------
__global__ void k_zero16(uint4* __restrict__ p, long n4) {
  long i = (long)blockIdx.x * blockDim.x + threadIdx.x;
  if (i < n4) { uint4 z; z.x = z.y = z.z = z.w = 0u; p[i] = z; }
}

// ------------------- a[t][v][w] -> bf16 a^T[t][w][v] ------------------------
__global__ void k_transpose_a(const float* __restrict__ a, __bf16* __restrict__ abT) {
  __shared__ __bf16 tile[32][33];
  int t  = blockIdx.z;
  int v0 = blockIdx.y * 32;
  int w0 = blockIdx.x * 32;
  int lx = threadIdx.x & 31;
  int ly = threadIdx.x >> 5;   // 0..7
  const float* ap = a + (size_t)t * V_D * V_D;
  #pragma unroll
  for (int i = 0; i < 32; i += 8)
    tile[ly + i][lx] = (__bf16)ap[(size_t)(v0 + ly + i) * V_D + (w0 + lx)];
  __syncthreads();
  __bf16* op = abT + (size_t)t * V_D * V_D;
  #pragma unroll
  for (int i = 0; i < 32; i += 8)
    op[(size_t)(w0 + ly + i) * V_D + (v0 + lx)] = tile[lx][ly + i];
}

// ----------- gcn 1x1 conv -> bf16 A-matrix xg[t][n*5+c][v] ------------------
__global__ void k_gcn(const float* __restrict__ vin, const float* __restrict__ gw,
                      const float* __restrict__ gb, __bf16* __restrict__ xg) {
  size_t id = (size_t)blockIdx.x * blockDim.x + threadIdx.x;   // 8*1280*1024
  int vv = (int)(id & (V_D - 1));
  size_t q = id >> 10;
  int r = (int)(q % ROWS);
  int t = (int)(q / ROWS);
  int n = r / C_OUT, c = r % C_OUT;
  float x = vin[(((size_t)n * 2 + 0) * T_D + t) * V_D + vv] * gw[c * 2 + 0]
          + vin[(((size_t)n * 2 + 1) * T_D + t) * V_D + vv] * gw[c * 2 + 1]
          + gb[c];
  xg[id] = (__bf16)x;
}

// --------------- einsum GEMM: y[r][t][w] = xg[t][r][:] @ a[t] ---------------
// one wave = 16 rows x 64 cols; software-pipelined k-loop of
// v_wmma_f32_16x16x32_bf16 (next-step fragments prefetched before WMMAs).
__global__ void __launch_bounds__(256)
k_einsum_gemm(const __bf16* __restrict__ xg,
              const __bf16* __restrict__ abT,
              float* __restrict__ y) {
  int wv   = (int)(((size_t)blockIdx.x * blockDim.x + threadIdx.x) >> 5); // 0..10239
  int lane = threadIdx.x & 31;
  int half = lane >> 4, lr = lane & 15;
  int t   = wv / 1280;
  int rem = wv % 1280;
  int r0  = (rem >> 4) << 4;     // row tile
  int c0  = (rem & 15) << 6;     // 64-col strip
  const __bf16* Ab = xg + ((size_t)t * ROWS + (r0 + lr)) * V_D;
  const __bf16* Bb = abT + (size_t)t * V_D * V_D + (size_t)(c0 + lr) * V_D;
  int ka = half * 16;            // A: lanes16-31 hold K 16..31
  int kb = half * 16;            // B: lanes16-31 hold K 16..31
  v8f acc0 = {}, acc1 = {}, acc2 = {}, acc3 = {};

  // prologue: fragments for k=0
  v16bf a  = ld16bf(Ab + ka, Ab + ka + 8);
  v16bf b0 = ld16bf(Bb + 0 * 16 * V_D + kb, Bb + 0 * 16 * V_D + kb + 8);
  v16bf b1 = ld16bf(Bb + 1 * 16 * V_D + kb, Bb + 1 * 16 * V_D + kb + 8);
  v16bf b2 = ld16bf(Bb + 2 * 16 * V_D + kb, Bb + 2 * 16 * V_D + kb + 8);
  v16bf b3 = ld16bf(Bb + 3 * 16 * V_D + kb, Bb + 3 * 16 * V_D + kb + 8);

  for (int k0 = 32; k0 < V_D; k0 += 32) {
    // prefetch next k-step while current WMMAs execute
    v16bf an  = ld16bf(Ab + k0 + ka, Ab + k0 + ka + 8);
    v16bf bn0 = ld16bf(Bb + 0 * 16 * V_D + k0 + kb, Bb + 0 * 16 * V_D + k0 + kb + 8);
    v16bf bn1 = ld16bf(Bb + 1 * 16 * V_D + k0 + kb, Bb + 1 * 16 * V_D + k0 + kb + 8);
    v16bf bn2 = ld16bf(Bb + 2 * 16 * V_D + k0 + kb, Bb + 2 * 16 * V_D + k0 + kb + 8);
    v16bf bn3 = ld16bf(Bb + 3 * 16 * V_D + k0 + kb, Bb + 3 * 16 * V_D + k0 + kb + 8);
    acc0 = __builtin_amdgcn_wmma_f32_16x16x32_bf16(false, a, false, b0, (short)0, acc0, false, false);
    acc1 = __builtin_amdgcn_wmma_f32_16x16x32_bf16(false, a, false, b1, (short)0, acc1, false, false);
    acc2 = __builtin_amdgcn_wmma_f32_16x16x32_bf16(false, a, false, b2, (short)0, acc2, false, false);
    acc3 = __builtin_amdgcn_wmma_f32_16x16x32_bf16(false, a, false, b3, (short)0, acc3, false, false);
    a = an; b0 = bn0; b1 = bn1; b2 = bn2; b3 = bn3;
  }
  // epilogue: last k-step
  acc0 = __builtin_amdgcn_wmma_f32_16x16x32_bf16(false, a, false, b0, (short)0, acc0, false, false);
  acc1 = __builtin_amdgcn_wmma_f32_16x16x32_bf16(false, a, false, b1, (short)0, acc1, false, false);
  acc2 = __builtin_amdgcn_wmma_f32_16x16x32_bf16(false, a, false, b2, (short)0, acc2, false, false);
  acc3 = __builtin_amdgcn_wmma_f32_16x16x32_bf16(false, a, false, b3, (short)0, acc3, false, false);

  int col = c0 + lr;
  #pragma unroll
  for (int i = 0; i < 8; ++i) {
    int row = r0 + i + half * 8;                 // C/D layout: VGPR i -> M=i / M=i+8
    float* yp = y + ((size_t)row * T_D + t) * V_D + col;
    yp[0] = acc0[i]; yp[16] = acc1[i]; yp[32] = acc2[i]; yp[48] = acc3[i];
  }
}

// ---- fused: BN1+PReLU -> (3,1) tcn -> BN2 + (res 1x1+BN from v) -> PReLU ---
// writes torch-.view-reinterpreted result into padded bf16 Xping
__global__ void k_tcn_fuse(const float* __restrict__ y, const float* __restrict__ vin,
                           const float* __restrict__ tw, const float* __restrict__ tb,
                           const float* __restrict__ b1g, const float* __restrict__ b1b,
                           const float* __restrict__ a1p,
                           const float* __restrict__ b2g, const float* __restrict__ b2b,
                           const float* __restrict__ rw, const float* __restrict__ rb,
                           const float* __restrict__ brg, const float* __restrict__ brb,
                           const float* __restrict__ a2p, __bf16* __restrict__ Xout) {
  size_t id = (size_t)blockIdx.x * blockDim.x + threadIdx.x;   // N*5*8*1024
  int w = (int)(id & (V_D - 1));
  size_t q = id >> 10;
  int t = (int)(q & 7); q >>= 3;
  int c = (int)(q % C_OUT);
  int n = (int)(q / C_OUT);
  const float inv = rsqrtf(1.0f + 1e-5f);
  float a1 = a1p[0];
  float u = tb[c];
  #pragma unroll
  for (int ci = 0; ci < C_OUT; ++ci) {
    float s1 = b1g[ci] * inv, sb = b1b[ci];
    #pragma unroll
    for (int dt = 0; dt < 3; ++dt) {
      int tt = t + dt - 1;
      if (tt < 0 || tt >= T_D) continue;
      float z = y[(((size_t)n * C_OUT + ci) * T_D + tt) * V_D + w] * s1 + sb;
      z = z >= 0.f ? z : a1 * z;
      u += z * tw[(c * C_OUT + ci) * 3 + dt];
    }
  }
  float z = u * (b2g[c] * inv) + b2b[c];
  float v0 = vin[(((size_t)n * 2 + 0) * T_D + t) * V_D + w];
  float v1 = vin[(((size_t)n * 2 + 1) * T_D + t) * V_D + w];
  float rs = (v0 * rw[c * 2 + 0] + v1 * rw[c * 2 + 1] + rb[c]) * (brg[c] * inv) + brb[c];
  z += rs;
  float a2 = a2p[0];
  z = z >= 0.f ? z : a2 * z;
  int lin = c * T_D + t;                 // flat (c,t) -> view (ch,h): ch*5+h
  int ch = lin / 5, hh = lin % 5;
  Xout[(size_t)n * N_STR + (size_t)ch * CH_STR + (size_t)(hh + 1) * XW + (w + 1)] = (__bf16)z;
}

// --------- pack all txpcnn weights into bf16 [stage][16][128] (K zero-pad) ---
__global__ void k_pack_w(const float* __restrict__ tp0_w, const float* __restrict__ tpk_w,
                         const float* __restrict__ tpout_w, __bf16* __restrict__ Wp) {
  int id = blockIdx.x * blockDim.x + threadIdx.x;   // 5*16*128
  int k  = id & 127;
  int co = (id >> 7) & 15;
  int s  = id >> 11;
  int ci = k / 9, r = k % 9;
  float w = 0.f;
  if (co < PT) {
    if (s == 0)      { if (ci < 8)  w = tp0_w[(co * 8 + ci) * 9 + r]; }
    else if (s <= 3) { if (ci < PT) w = tpk_w[((((s - 1) * PT) + co) * PT + ci) * 9 + r]; }
    else             { if (ci < PT) w = tpout_w[(co * PT + ci) * 9 + r]; }
  }
  Wp[id] = (__bf16)w;
}

// ----------------- txpcnn 3x3 conv as implicit-GEMM WMMA --------------------
// one wave = 16 output channels x 16 pixels at fixed (n,h); K=128 (zero-pad).
// All 4 B-fragment gathers + 4 A-fragments issued before the WMMA burst so
// the 64 u16 gather loads overlap instead of serializing per k-step.
__global__ void __launch_bounds__(256)
k_txp_conv(const __bf16* __restrict__ Xin, const __bf16* __restrict__ Wp,
           const float* __restrict__ bias, const float* __restrict__ alp,
           int aidx, __bf16* __restrict__ Xout, float* __restrict__ Fout,
           int do_prelu, int do_res) {
  __shared__ int offs[128];
  for (int k = threadIdx.x; k < 128; k += blockDim.x) {
    int ci = k / 9, r = k % 9, dh = r / 3, dw = r % 3;
    offs[k] = (ci < XC) ? (ci * CH_STR + dh * XW + dw) : 0;   // zero-weight pad lanes
  }
  __syncthreads();
  int wv   = (int)(((size_t)blockIdx.x * blockDim.x + threadIdx.x) >> 5); // 0..81919
  int lane = threadIdx.x & 31;
  int half = lane >> 4, lr = lane & 15;
  int w0 = (wv & 63) << 4;
  int h  = (wv >> 6) % 5;
  int n  = wv / 320;
  const __bf16* pb = Xin + (size_t)n * N_STR + (size_t)h * XW + (w0 + lr);
  const __bf16* wp = Wp + lr * 128;

  v16bf afr[4];
  union { v16bf v; __bf16 e[16]; } bu[4];
  #pragma unroll
  for (int s = 0; s < 4; ++s) {
    int k0 = s * 32;
    afr[s] = ld16bf(wp + k0 + half * 8, wp + k0 + half * 8 + 16);
    int kb = k0 + half * 16;
    #pragma unroll
    for (int j = 0; j < 16; ++j) bu[s].e[j] = pb[offs[k0 + half * 8 + (j & 7) + (j >> 3) * 16]];
  }
  v8f acc = {};
  #pragma unroll
  for (int s = 0; s < 4; ++s)
    acc = __builtin_amdgcn_wmma_f32_16x16x32_bf16(false, afr[s], false, bu[s].v,
                                                  (short)0, acc, false, false);

  float alpha = do_prelu ? alp[aidx] : 0.f;
  #pragma unroll
  for (int i = 0; i < 8; ++i) {
    int co = i + half * 8;
    if (co < PT) {
      float val = acc[i] + bias[co];
      if (do_prelu) val = val >= 0.f ? val : alpha * val;
      if (do_res)
        val += (float)Xin[(size_t)n * N_STR + (size_t)co * CH_STR + (size_t)(h + 1) * XW + (w0 + lr + 1)];
      if (Fout)   // final stage: flat (N,12,5,1024) == output .view, f32
        Fout[(((size_t)n * PT + co) * 5 + h) * V_D + w0 + lr] = val;
      else
        Xout[(size_t)n * N_STR + (size_t)co * CH_STR + (size_t)(h + 1) * XW + (w0 + lr + 1)] = (__bf16)val;
    }
  }
}

// ---------------------------------------------------------------------------
extern "C" void kernel_launch(void* const* d_in, const int* in_sizes, int n_in,
                              void* d_out, int out_size, void* d_ws, size_t ws_size,
                              hipStream_t stream) {
  const float* v       = (const float*)d_in[0];
  const float* a       = (const float*)d_in[1];
  const float* gcn_w   = (const float*)d_in[2];
  const float* gcn_b   = (const float*)d_in[3];
  const float* bn1_g   = (const float*)d_in[4];
  const float* bn1_b   = (const float*)d_in[5];
  const float* prelu1  = (const float*)d_in[6];
  const float* tcn_w   = (const float*)d_in[7];
  const float* tcn_b   = (const float*)d_in[8];
  const float* bn2_g   = (const float*)d_in[9];
  const float* bn2_b   = (const float*)d_in[10];
  const float* res_w   = (const float*)d_in[11];
  const float* res_b   = (const float*)d_in[12];
  const float* bnr_g   = (const float*)d_in[13];
  const float* bnr_b   = (const float*)d_in[14];
  const float* prelu2  = (const float*)d_in[15];
  const float* tp0_w   = (const float*)d_in[16];
  const float* tp0_b   = (const float*)d_in[17];
  const float* tpk_w   = (const float*)d_in[18];
  const float* tpk_b   = (const float*)d_in[19];
  const float* tpout_w = (const float*)d_in[20];
  const float* tpout_b = (const float*)d_in[21];
  const float* prelu_tp= (const float*)d_in[22];

  char* ws = (char*)d_ws;
  __bf16* abT   = (__bf16*)(ws + OFF_ABT);
  __bf16* xg    = (__bf16*)(ws + OFF_XG);
  float*  y     = (float*)(ws + OFF_Y);
  __bf16* Xping = (__bf16*)(ws + OFF_XPING);
  __bf16* Xpong = (__bf16*)(ws + OFF_XPONG);
  __bf16* Wp    = (__bf16*)(ws + OFF_WP);

  float* out   = (float*)d_out;
  float* out_a = out + (size_t)N_B * C_OUT * PT * V_D;   // tail: echo of a

  const long x4 = (long)(XBUF_BYTES / 16);
  const int  ZB = (int)((x4 + 255) / 256);

  // zero Xping halos (interior rewritten every call)
  k_zero16<<<ZB, 256, 0, stream>>>((uint4*)Xping, x4);
  // pack txpcnn weights (bf16, K zero-padded to 128)
  k_pack_w<<<40, 256, 0, stream>>>(tp0_w, tpk_w, tpout_w, Wp);
  // a -> bf16 a^T
  k_transpose_a<<<dim3(32, 32, 8), 256, 0, stream>>>(a, abT);
  // gcn pointwise -> bf16 A matrix
  k_gcn<<<40960, 256, 0, stream>>>(v, gcn_w, gcn_b, xg);
  // 8x (1280 x 1024 x 1024) bf16 WMMA GEMM
  k_einsum_gemm<<<1280, 256, 0, stream>>>(xg, abT, y);
  // fused BN/PReLU/tcn/BN+res/PReLU -> padded Xping (y, abT, xg now dead)
  k_tcn_fuse<<<40960, 256, 0, stream>>>(y, v, tcn_w, tcn_b, bn1_g, bn1_b, prelu1,
                                        bn2_g, bn2_b, res_w, res_b, bnr_g, bnr_b,
                                        prelu2, Xping);
  // zero Xpong (aliases dead GEMM buffers)
  k_zero16<<<ZB, 256, 0, stream>>>((uint4*)Xpong, x4);

  // txpcnn stages (WMMA implicit GEMM): s0 ping->pong, s1..s3 residual ping-pong,
  // s4 ping -> f32 d_out
  k_txp_conv<<<10240, 256, 0, stream>>>(Xping, Wp + 0 * 2048, tp0_b, prelu_tp, 0,
                                        Xpong, nullptr, 1, 0);
  k_txp_conv<<<10240, 256, 0, stream>>>(Xpong, Wp + 1 * 2048, tpk_b + 0 * PT, prelu_tp, 1,
                                        Xping, nullptr, 1, 1);
  k_txp_conv<<<10240, 256, 0, stream>>>(Xping, Wp + 2 * 2048, tpk_b + 1 * PT, prelu_tp, 2,
                                        Xpong, nullptr, 1, 1);
  k_txp_conv<<<10240, 256, 0, stream>>>(Xpong, Wp + 3 * 2048, tpk_b + 2 * PT, prelu_tp, 3,
                                        Xping, nullptr, 1, 1);
  k_txp_conv<<<10240, 256, 0, stream>>>(Xping, Wp + 4 * 2048, tpout_b, prelu_tp, 0,
                                        nullptr, out, 0, 0);

  // second tuple output: a (f32) copied verbatim
  hipMemcpyAsync(out_a, a, (size_t)T_D * V_D * V_D * sizeof(float),
                 hipMemcpyDeviceToDevice, stream);
}
